// MILModel_46497315947049
// MI455X (gfx1250) — matrix operand order repaired and checked by reading, not compiled, not run on hardware
//
#include <hip/hip_runtime.h>
#include <math.h>

// ---------------- problem constants ----------------
#define BB   16
#define NN   2048
#define DD   1024
#define HH   256
#define AA   128
#define EPSL 1e-5f
#define NEGV -1e9f

typedef __attribute__((ext_vector_type(16))) __bf16 v16bf;
typedef __attribute__((ext_vector_type(8)))  float  v8f;

#define LDA 40            // LDS tile stride (bf16 units): 80B -> 16B aligned for async B128,
                          // 20-dword stride is conflict-free across 16-lane fragment groups

// ---------- CDNA5 async-copy path (ASYNCcnt-tracked, ISA 08 §4) ----------
__device__ __forceinline__ void async_load_b128(unsigned lds_off, const void* gaddr) {
    asm volatile("global_load_async_to_lds_b128 %0, %1, off"
                 :: "v"(lds_off), "v"(gaddr) : "memory");
}
__device__ __forceinline__ void wait_async0() {
    asm volatile("s_wait_asynccnt 0" ::: "memory");
}
__device__ __forceinline__ unsigned lds_addr(const void* p) {
    // generic flat addr of LDS = {shared aperture, 32-bit LDS offset}; truncate
    return (unsigned)(size_t)p;
}

// ===================================================================
// Kernel P: convert W1 (256x1024) and Wa1 (128x256) to bf16
// ===================================================================
__global__ void __launch_bounds__(256) convert_weights(
    const float* __restrict__ W1, const float* __restrict__ Wa1,
    __bf16* __restrict__ W1b, __bf16* __restrict__ Wa1b)
{
    const int n1 = HH * DD;
    const int n2 = AA * HH;
    int i = blockIdx.x * blockDim.x + threadIdx.x;
    if (i < n1)            W1b[i]       = (__bf16)W1[i];
    else if (i < n1 + n2)  Wa1b[i - n1] = (__bf16)Wa1[i - n1];
}

// ---------- fragment loaders (ISA 7.12.2 layouts, wave32) ----------
__device__ __forceinline__ v16bf load_a_frag(const __bf16* ap, int half) {
    v16bf a;
#pragma unroll
    for (int r = 0; r < 4; ++r) {
        int k = 2 * r + 8 * half;
        a[2 * r] = ap[k]; a[2 * r + 1] = ap[k + 1];
    }
#pragma unroll
    for (int r = 4; r < 8; ++r) {
        int k = 16 + 2 * (r - 4) + 8 * half;
        a[2 * r] = ap[k]; a[2 * r + 1] = ap[k + 1];
    }
    return a;
}
__device__ __forceinline__ v16bf load_b_frag(const __bf16* bp, int half) {
    v16bf b;
    const __bf16* p = bp + 8 * half;
#pragma unroll
    for (int r = 0; r < 8; ++r) { int k = 2 * r + (r >= 4 ? 8 : 0); b[2 * r] = p[k]; b[2 * r + 1] = p[k + 1]; }
    return b;
}

// ===================================================================
// Kernel A: h = GELU(LN(x @ W1^T + b1)); scores = h . Wc + bc
// 256 blocks x 256 thr. Block: 128 rows. Wave: 16 rows x 256 cols.
// Double-buffered K-loop: A tile manual (f32->bf16 fused convert),
// B tile via global_load_async_to_lds_b128.
// ===================================================================
#define ATILE (128 * LDA * 2)          // bytes per A buffer  (10240)
#define BTILE (256 * LDA * 2)          // bytes per B buffer  (20480)

__global__ void __launch_bounds__(256) mil_gemm1(
    const float* __restrict__ x, const __bf16* __restrict__ W1b,
    const float* __restrict__ b1, const float* __restrict__ ln_g,
    const float* __restrict__ ln_b, const float* __restrict__ Wc,
    const float* __restrict__ bc,
    __bf16* __restrict__ hb, float* __restrict__ scores)
{
    __shared__ unsigned char smem[2 * ATILE + 2 * BTILE];   // 61440 B (consts overlay after loop)

    const int tid  = threadIdx.x;
    const int lane = tid & 31;
    const int w    = tid >> 5;
    const int half = lane >> 4;
    const int l16  = lane & 15;
    const int m0   = blockIdx.x * 128;

    __bf16* As = (__bf16*)smem;                       // 2 buffers of 128xLDA
    __bf16* Bs = (__bf16*)(smem + 2 * ATILE);         // 2 buffers of 256xLDA
    const unsigned bsBase = lds_addr(Bs);
    const char* w1bBytes  = (const char*)W1b;

    const float bc0 = bc[0];

    v8f acc[16];
#pragma unroll
    for (int nt = 0; nt < 16; ++nt) acc[nt] = (v8f){0,0,0,0,0,0,0,0};

    const int arow = tid >> 1;
    const int acol = (tid & 1) * 16;
    const float* xrow = x + (size_t)(m0 + arow) * DD + acol;

    // ---- tile loaders ----
    auto loadA = [&](int kt, int buf) {
        const int k0 = kt * 32;
#pragma unroll
        for (int j = 0; j < 16; j += 4) {
            float4 v = *(const float4*)(xrow + k0 + j);
            __bf16* d = &As[buf * 128 * LDA + arow * LDA + acol + j];
            d[0] = (__bf16)v.x; d[1] = (__bf16)v.y;
            d[2] = (__bf16)v.z; d[3] = (__bf16)v.w;
        }
        __builtin_prefetch(xrow + k0 + 32, 0, 0);   // global_prefetch_b8 (next x tile)
    };
    auto loadB = [&](int kt, int buf) {
        const int k0 = kt * 32;
#pragma unroll
        for (int is = 0; is < 4; ++is) {
            int idx = is * 256 + tid;      // 0..1023
            int n = idx >> 2;              // col row 0..255
            int q = idx & 3;               // 16B chunk
            async_load_b128(bsBase + (unsigned)(buf * BTILE + n * (LDA * 2) + q * 16),
                            w1bBytes + (size_t)n * (DD * 2) + (size_t)k0 * 2 + q * 16);
        }
    };

    loadA(0, 0); loadB(0, 0);

    for (int kt = 0; kt < 32; ++kt) {
        const int cur = kt & 1;
        wait_async0();                 // buf[cur] async writes complete (this wave)
        __syncthreads();               // dscnt flush + visibility across waves
        if (kt + 1 < 32) { loadA(kt + 1, cur ^ 1); loadB(kt + 1, cur ^ 1); }

        const v16bf afrag = load_a_frag(&As[cur * 128 * LDA + (w * 16 + l16) * LDA], half);
#pragma unroll
        for (int nt = 0; nt < 16; ++nt) {
            const v16bf bfrag = load_b_frag(&Bs[cur * 256 * LDA + (nt * 16 + l16) * LDA], half);
            acc[nt] = __builtin_amdgcn_wmma_f32_16x16x32_bf16(
                false, afrag, false, bfrag, (short)0, acc[nt], false, false);
        }
    }

    // ---- constants overlay (tiles dead now) ----
    __syncthreads();
    float* cb1 = (float*)smem;
    float* cgv = cb1 + HH;
    float* cbt = cgv + HH;
    float* cwc = cbt + HH;
    cb1[tid] = b1[tid]; cgv[tid] = ln_g[tid]; cbt[tid] = ln_b[tid]; cwc[tid] = Wc[tid];
    __syncthreads();

    // ---- epilogue: bias + LayerNorm + GELU + score GEMV, in-register ----
    // acc[nt][r] = element (row = w*16 + r + 8*half, col = nt*16 + l16)
#pragma unroll
    for (int r = 0; r < 8; ++r) {
        float s = 0.f, sq = 0.f;
        float v[16];
#pragma unroll
        for (int nt = 0; nt < 16; ++nt) {
            float t = acc[nt][r] + cb1[nt * 16 + l16];
            v[nt] = t; s += t; sq += t * t;
        }
#pragma unroll
        for (int off = 1; off < 16; off <<= 1) {
            s  += __shfl_xor(s,  off);
            sq += __shfl_xor(sq, off);
        }
        const float mu  = s * (1.0f / HH);
        const float var = sq * (1.0f / HH) - mu * mu;
        const float rs  = rsqrtf(var + EPSL);
        const int rowG  = m0 + w * 16 + r + 8 * half;

        float scp = 0.f;
#pragma unroll
        for (int nt = 0; nt < 16; ++nt) {
            const int col = nt * 16 + l16;
            float n  = (v[nt] - mu) * rs * cgv[col] + cbt[col];
            float ge = 0.5f * n * (1.0f + erff(n * 0.70710678118654752f));  // exact GELU
            hb[(size_t)rowG * HH + col] = (__bf16)ge;
            scp += ge * cwc[col];
        }
#pragma unroll
        for (int off = 1; off < 16; off <<= 1) scp += __shfl_xor(scp, off);
        if (l16 == 0) scores[rowG] = scp + bc0;
    }
}

// ===================================================================
// Kernel B: logits = tanh(h @ Wa1^T + ba1) . Wa2 + ba2
// 256 blocks x 256 thr. Block: 128 rows. Wave: 16 rows x 128 cols.
// Both tiles via async-to-LDS, double-buffered.
// ===================================================================
__global__ void __launch_bounds__(256) mil_attn(
    const __bf16* __restrict__ hb, const __bf16* __restrict__ Wa1b,
    const float* __restrict__ ba1, const float* __restrict__ Wa2,
    const float* __restrict__ ba2, float* __restrict__ logits)
{
    __shared__ unsigned char smem[4 * ATILE];       // A:2 buf, B:2 buf (40960 B)
    __shared__ float cba1[AA];

    const int tid  = threadIdx.x;
    const int lane = tid & 31;
    const int w    = tid >> 5;
    const int half = lane >> 4;
    const int l16  = lane & 15;
    const int m0   = blockIdx.x * 128;

    __bf16* As = (__bf16*)smem;                     // 2 x 128 x LDA
    __bf16* Bs = (__bf16*)(smem + 2 * ATILE);       // 2 x 128 x LDA
    const unsigned asBase = lds_addr(As);
    const unsigned bsBase = lds_addr(Bs);
    const char* hbBytes  = (const char*)(hb + (size_t)m0 * HH);
    const char* wa1Bytes = (const char*)Wa1b;

    if (tid < AA) cba1[tid] = ba1[tid];
    const float ba2v = ba2[0];

    float wa2v[8];
#pragma unroll
    for (int nt = 0; nt < 8; ++nt) wa2v[nt] = Wa2[nt * 16 + l16];

    v8f acc[8];
#pragma unroll
    for (int nt = 0; nt < 8; ++nt) acc[nt] = (v8f){0,0,0,0,0,0,0,0};

    auto loadTile = [&](const char* src, unsigned dstBase, int kt, int buf) {
        const int k0 = kt * 32;
#pragma unroll
        for (int is = 0; is < 2; ++is) {
            int idx = is * 256 + tid;   // 0..511
            int n = idx >> 2;           // 0..127
            int q = idx & 3;
            async_load_b128(dstBase + (unsigned)(buf * ATILE + n * (LDA * 2) + q * 16),
                            src + (size_t)n * (HH * 2) + (size_t)k0 * 2 + q * 16);
        }
    };

    loadTile(hbBytes, asBase, 0, 0);
    loadTile(wa1Bytes, bsBase, 0, 0);

    for (int kt = 0; kt < 8; ++kt) {
        const int cur = kt & 1;
        wait_async0();
        __syncthreads();
        if (kt + 1 < 8) {
            loadTile(hbBytes,  asBase, kt + 1, cur ^ 1);
            loadTile(wa1Bytes, bsBase, kt + 1, cur ^ 1);
        }
        const v16bf afrag = load_a_frag(&As[cur * 128 * LDA + (w * 16 + l16) * LDA], half);
#pragma unroll
        for (int nt = 0; nt < 8; ++nt) {
            const v16bf bfrag = load_b_frag(&Bs[cur * 128 * LDA + (nt * 16 + l16) * LDA], half);
            acc[nt] = __builtin_amdgcn_wmma_f32_16x16x32_bf16(
                false, afrag, false, bfrag, (short)0, acc[nt], false, false);
        }
    }

    // tanh + Wa2 GEMV + ba2
#pragma unroll
    for (int r = 0; r < 8; ++r) {
        float p = 0.f;
#pragma unroll
        for (int nt = 0; nt < 8; ++nt) {
            float t = tanhf(acc[nt][r] + cba1[nt * 16 + l16]);
            p += t * wa2v[nt];
        }
#pragma unroll
        for (int off = 1; off < 16; off <<= 1) p += __shfl_xor(p, off);
        if (l16 == 0) logits[m0 + w * 16 + r + 8 * half] = p + ba2v;
    }
}

// ===================================================================
// Kernel C: per-sample masked softmax pooling + top-k mean (bitonic)
// ===================================================================
__global__ void __launch_bounds__(256) mil_pool(
    const float* __restrict__ scores, const float* __restrict__ logits,
    const int* __restrict__ lengths, float* __restrict__ video)
{
    __shared__ float sarr[NN];
    __shared__ float red[8];

    const int b   = blockIdx.x;
    const int tid = threadIdx.x;
    const int w   = tid >> 5;
    const int T   = lengths[b];
    const float* sc = scores + (size_t)b * NN;
    const float* lg = logits + (size_t)b * NN;

    // masked max of logits
    float mx = -INFINITY;
    for (int i = tid; i < NN; i += 256) if (i < T) mx = fmaxf(mx, lg[i]);
#pragma unroll
    for (int off = 1; off < 32; off <<= 1) mx = fmaxf(mx, __shfl_xor(mx, off));
    if ((tid & 31) == 0) red[w] = mx;
    __syncthreads();
    if (tid == 0) { float m = red[0]; for (int i = 1; i < 8; ++i) m = fmaxf(m, red[i]); red[0] = m; }
    __syncthreads();
    mx = red[0];

    // exp-sum, weighted score sum; fill sort array
    float se = 0.f, sw = 0.f;
    for (int i = tid; i < NN; i += 256) {
        float s = sc[i];
        if (i < T) { float e = __expf(lg[i] - mx); se += e; sw += e * s; sarr[i] = s; }
        else sarr[i] = NEGV;
    }
#pragma unroll
    for (int off = 1; off < 32; off <<= 1) { se += __shfl_xor(se, off); sw += __shfl_xor(sw, off); }
    __syncthreads();
    if ((tid & 31) == 0) red[w] = se;
    __syncthreads();
    float seT = 0.f; if (tid == 0) { for (int i = 0; i < 8; ++i) seT += red[i]; red[0] = seT; }
    __syncthreads();
    seT = red[0];
    __syncthreads();
    if ((tid & 31) == 0) red[w] = sw;
    __syncthreads();
    float swT = 0.f; if (tid == 0) { for (int i = 0; i < 8; ++i) swT += red[i]; red[0] = swT; }
    __syncthreads();
    swT = red[0];

    // bitonic sort descending
    for (int size = 2; size <= NN; size <<= 1) {
        for (int stride = size >> 1; stride > 0; stride >>= 1) {
            __syncthreads();
            for (int c = tid; c < NN / 2; c += 256) {
                int i = 2 * c - (c & (stride - 1));
                int j = i + stride;
                float a = sarr[i], bb = sarr[j];
                bool desc = ((i & size) == 0);
                if (desc ? (a < bb) : (a > bb)) { sarr[i] = bb; sarr[j] = a; }
            }
        }
    }
    __syncthreads();

    int k = T / 10; if (k < 1) k = 1;
    float ts = 0.f;
    for (int i = tid; i < k; i += 256) ts += sarr[i];
#pragma unroll
    for (int off = 1; off < 32; off <<= 1) ts += __shfl_xor(ts, off);
    __syncthreads();
    if ((tid & 31) == 0) red[w] = ts;
    __syncthreads();
    if (tid == 0) {
        float tsum = 0.f; for (int i = 0; i < 8; ++i) tsum += red[i];
        float attn = (T > 0) ? (swT / seT) : 0.f;
        video[b] = (T > 0) ? 0.5f * attn + 0.5f * (tsum / (float)k) : 0.f;
    }
}

// ===================================================================
extern "C" void kernel_launch(void* const* d_in, const int* in_sizes, int n_in,
                              void* d_out, int out_size, void* d_ws, size_t ws_size,
                              hipStream_t stream) {
    (void)in_sizes; (void)n_in; (void)out_size; (void)ws_size;
    const float* x    = (const float*)d_in[0];
    const int*   lens = (const int*)  d_in[1];
    const float* W1   = (const float*)d_in[2];
    const float* b1   = (const float*)d_in[3];
    const float* ln_g = (const float*)d_in[4];
    const float* ln_b = (const float*)d_in[5];
    const float* Wa1  = (const float*)d_in[6];
    const float* ba1  = (const float*)d_in[7];
    const float* Wa2  = (const float*)d_in[8];
    const float* ba2  = (const float*)d_in[9];
    const float* Wc   = (const float*)d_in[10];
    const float* bc   = (const float*)d_in[11];
    float* out = (float*)d_out;   // [0..15]=video, [16..]=instance_scores

    char* ws = (char*)d_ws;
    __bf16* W1b    = (__bf16*)(ws);                 // 512 KB
    __bf16* Wa1b   = (__bf16*)(ws + 524288);        //  64 KB
    __bf16* hb     = (__bf16*)(ws + 589824);        //  16 MB
    float*  logits = (float*) (ws + 17367040);      // 128 KB

    convert_weights<<<1152, 256, 0, stream>>>(W1, Wa1, W1b, Wa1b);
    mil_gemm1<<<256, 256, 0, stream>>>(x, W1b, b1, ln_g, ln_b, Wc, bc, hb, out + BB);
    mil_attn <<<256, 256, 0, stream>>>(hb, Wa1b, ba1, Wa2, ba2, logits);
    mil_pool <<<BB, 256, 0, stream>>>(out + BB, logits, lens, out);
}
